// LinkPredictor_61512521613946
// MI455X (gfx1250) — matrix operand order repaired and checked by reading, not compiled, not run on hardware
//
#include <hip/hip_runtime.h>
#include <cmath>

typedef __attribute__((ext_vector_type(16))) _Float16 v16h;
typedef __attribute__((ext_vector_type(8)))  float    v8f;
typedef __attribute__((ext_vector_type(4)))  float    v4f;
typedef _Float16 f16;

#define EM 64        // edges per block
#define TPB 256
#define NWAVES 8
#define ZOUT 768
#define DD 256

// LDS carve (bytes):
//  PROD  64*768 f16 = 98304
//  RAW   64*768 f16 = 98304
//  AB    64*256 f16 = 32768
//  Q     64*384 f16 = 49152
//  rxz   64*12  f32 =  3072
//  stats 64*8   f32 =  2048
//  sidx  128    i32 =   512
static constexpr int SMEM_BYTES = 98304 + 98304 + 32768 + 49152 + 3072 + 2048 + 512; // 284160

__device__ __forceinline__ float silu_f(float v) { return v / (1.f + __expf(-v)); }

union Frag16 { v16h h; v4f f[2]; };

// A fragment (16x32 f16, ISA 7.12.2): lane m = lane&15; lanes<16 hold K {k..k+7, k+16..k+23},
// lanes>=16 hold K {k+8..k+15, k+24..k+31}. Row-major LDS [rows][strideK] halves.
__device__ __forceinline__ v16h load_a_frag(const f16* A, int strideK, int mt, int kt, int lane) {
    const f16* base = A + (mt * 16 + (lane & 15)) * strideK + kt * 32 + ((lane >> 4) * 8);
    Frag16 fr;
    fr.f[0] = *reinterpret_cast<const v4f*>(base);
    fr.f[1] = *reinterpret_cast<const v4f*>(base + 16);
    return fr.h;
}

// B fragment (32x16 f16): lane holds output column n = nt*16 + (lane&15), K halves
// {k+8h..k+8h+7, k+16+8h..k+16+8h+7} with h = lane>>4 (same as A). Weights stored [out][in] f16 row-major, strideK halves.
__device__ __forceinline__ v16h load_b_frag(const f16* B, int strideK, int nt, int kt, int lane) {
    const f16* base = B + (nt * 16 + (lane & 15)) * strideK + kt * 32 + ((lane >> 4) * 8);
    Frag16 fr;
    fr.f[0] = *reinterpret_cast<const v4f*>(base);
    fr.f[1] = *reinterpret_cast<const v4f*>(base + 16);
    return fr.h;
}

// OUT[64 x (ntiles*16)] = A(64 x ktiles*32) * B^T, f16 output with optional bias+silu.
// 2x2 register tile blocking: each wave iterates 32x32 output blocks (2 A-frags + 2 B-frags
// feed 4 WMMAs per k-step -> halves fragment loads per WMMA vs 1x1 tiling).
__device__ __forceinline__ void gemm_f16(const f16* A, int sA, const f16* Bg, int sB,
                                         int ntiles, int ktiles,
                                         f16* out, int sOut,
                                         const float* bias, int do_silu, int tid) {
    const int lane = tid & 31, wave = tid >> 5;
    const int nblocks = 2 * (ntiles >> 1);   // (4 mtiles / 2) * (ntiles / 2)

    auto store_tile = [&](v8f acc, int mt, int nt) {
        const int col  = nt * 16 + (lane & 15);
        const int row0 = mt * 16 + ((lane >> 4) * 8);
        const float bv = bias ? bias[col] : 0.f;
#pragma unroll
        for (int v = 0; v < 8; ++v) {
            float val = acc[v] + bv;
            if (do_silu) val = silu_f(val);
            out[(row0 + v) * sOut + col] = (f16)val;
        }
    };

    for (int t = wave; t < nblocks; t += NWAVES) {
        const int mt0 = (t & 1) * 2;
        const int nt0 = (t >> 1) * 2;
        v8f a00 = {0.f, 0.f, 0.f, 0.f, 0.f, 0.f, 0.f, 0.f};
        v8f a01 = a00, a10 = a00, a11 = a00;
        for (int kt = 0; kt < ktiles; ++kt) {
            v16h af0 = load_a_frag(A, sA, mt0 + 0, kt, lane);
            v16h af1 = load_a_frag(A, sA, mt0 + 1, kt, lane);
            v16h bf0 = load_b_frag(Bg, sB, nt0 + 0, kt, lane);
            v16h bf1 = load_b_frag(Bg, sB, nt0 + 1, kt, lane);
            if (kt + 1 < ktiles)
                __builtin_prefetch(Bg + (nt0 * 16 + (lane & 15)) * sB + (kt + 1) * 32, 0, 1);
            a00 = __builtin_amdgcn_wmma_f32_16x16x32_f16(false, af0, false, bf0, (short)0, a00,
                                                         false, false);
            a01 = __builtin_amdgcn_wmma_f32_16x16x32_f16(false, af0, false, bf1, (short)0, a01,
                                                         false, false);
            a10 = __builtin_amdgcn_wmma_f32_16x16x32_f16(false, af1, false, bf0, (short)0, a10,
                                                         false, false);
            a11 = __builtin_amdgcn_wmma_f32_16x16x32_f16(false, af1, false, bf1, (short)0, a11,
                                                         false, false);
        }
        store_tile(a00, mt0 + 0, nt0 + 0);
        store_tile(a01, mt0 + 0, nt0 + 1);
        store_tile(a10, mt0 + 1, nt0 + 0);
        store_tile(a11, mt0 + 1, nt0 + 1);
    }
}

// LayerNorm over 768 cols of RAW (+bias, optional silu first, per reference), folded into PROD.
// mode 0: PROD = ln ; 1: PROD += ln ; 2: PROD *= ln
__device__ __forceinline__ void ln_apply(const f16* raw, f16* prod, float* stats,
                                         const float* bias, const float* g, const float* b,
                                         int act_silu, int mode, int tid) {
    const int e = tid >> 2, r = tid & 3;
    const int c0 = r * 192;
    float s = 0.f, s2 = 0.f;
    for (int i = 0; i < 192; ++i) {
        float v = (float)raw[e * ZOUT + c0 + i] + bias[c0 + i];
        if (act_silu) v = silu_f(v);
        s += v; s2 += v * v;
    }
    stats[e * 8 + r * 2 + 0] = s;
    stats[e * 8 + r * 2 + 1] = s2;
    __syncthreads();
    float ts = 0.f, ts2 = 0.f;
#pragma unroll
    for (int j = 0; j < 4; ++j) { ts += stats[e * 8 + j * 2]; ts2 += stats[e * 8 + j * 2 + 1]; }
    const float mean = ts * (1.f / 768.f);
    const float var  = ts2 * (1.f / 768.f) - mean * mean;
    const float rstd = __frsqrt_rn(var + 1e-5f);
    for (int i = 0; i < 192; ++i) {
        const int c = c0 + i;
        float v = (float)raw[e * ZOUT + c] + bias[c];
        if (act_silu) v = silu_f(v);
        const float ln = (v - mean) * rstd * g[c] + b[c];
        f16* p = &prod[e * ZOUT + c];
        if (mode == 0)      *p = (f16)ln;
        else if (mode == 1) *p = (f16)((float)*p + ln);
        else                *p = (f16)((float)*p * ln);
    }
    __syncthreads();
}

// rxz[e][c*3+d] (optionally transposed) -> silu(linear 9->256) f16 into AB
__device__ __forceinline__ void g1_pass(const float* rxzl, const float* z_w1, const float* z_b1,
                                        f16* AB, int transpose, int tid) {
    const int e = tid >> 2, r = tid & 3;
    float rv[9];
#pragma unroll
    for (int c = 0; c < 3; ++c)
#pragma unroll
        for (int d = 0; d < 3; ++d)
            rv[c * 3 + d] = transpose ? rxzl[e * 12 + d * 3 + c] : rxzl[e * 12 + c * 3 + d];
    for (int o = r * 64; o < r * 64 + 64; ++o) {
        float acc = z_b1[o];
#pragma unroll
        for (int k = 0; k < 9; ++k) acc += rv[k] * z_w1[o * 9 + k];
        AB[e * DD + o] = (f16)silu_f(acc);
    }
}

__global__ __launch_bounds__(TPB) void fused_edge_kernel(
    const float* __restrict__ x, const float* __restrict__ Z, const int* __restrict__ tar_ei,
    const float* __restrict__ xij_b1, const float* __restrict__ xij_b2,
    const float* __restrict__ xij_ln_g, const float* __restrict__ xij_ln_b,
    const float* __restrict__ z_w1, const float* __restrict__ z_b1,
    const float* __restrict__ z_b2, const float* __restrict__ z_ln_g,
    const float* __restrict__ z_ln_b,
    const float* __restrict__ zx_b, const float* __restrict__ zx_ln_g,
    const float* __restrict__ zx_ln_b,
    const float* __restrict__ lin_w, const float* __restrict__ lin_b,
    const f16* __restrict__ W1h, const f16* __restrict__ W2h, const f16* __restrict__ ZW2h,
    const f16* __restrict__ ZXWh, const f16* __restrict__ ZXth,
    float* __restrict__ out, int E_, int N_) {
    extern __shared__ char smem[];
    f16*   PROD  = (f16*)smem;                    // 64*768
    f16*   RAW   = PROD + EM * ZOUT;              // 64*768
    f16*   AB    = RAW + EM * ZOUT;               // 64*256
    f16*   Q     = AB + EM * DD;                  // 64*384
    float* rxz   = (float*)(Q + EM * 384);        // 64*12
    float* stats = rxz + EM * 12;                 // 64*8
    int*   sidx  = (int*)(stats + EM * 8);        // 128

    const int tid = threadIdx.x;
    const int e0  = blockIdx.x * EM;
    const int e   = tid >> 2, r = tid & 3;

    if (tid < 2 * EM) {
        const int which = tid >> 6;
        int ge = e0 + (tid & 63);
        if (ge >= E_) ge = E_ - 1;
        sidx[tid] = tar_ei[which * E_ + ge];
    }
    for (int i = tid; i < EM * 12; i += TPB) rxz[i] = 0.f;
    __syncthreads();

    // ---- Z pass: rxz (cross-dots) and Q = Zi .* Zj ----
    {
        const int si = sidx[e], di = sidx[EM + e];
        const float* Zi = Z + (size_t)si * 384 + r * 32;
        const float* Zj = Z + (size_t)di * 384 + r * 32;
        float pcd[9];
#pragma unroll
        for (int k = 0; k < 9; ++k) pcd[k] = 0.f;
        for (int l = 0; l < 32; ++l) {
            const float a0 = Zi[l], a1 = Zi[128 + l], a2 = Zi[256 + l];
            const float b0 = Zj[l], b1 = Zj[128 + l], b2 = Zj[256 + l];
            pcd[0] += a0 * b0; pcd[1] += a0 * b1; pcd[2] += a0 * b2;
            pcd[3] += a1 * b0; pcd[4] += a1 * b1; pcd[5] += a1 * b2;
            pcd[6] += a2 * b0; pcd[7] += a2 * b1; pcd[8] += a2 * b2;
            const int lg = r * 32 + l;
            Q[e * 384 + lg]       = (f16)(a0 * b0);
            Q[e * 384 + 128 + lg] = (f16)(a1 * b1);
            Q[e * 384 + 256 + lg] = (f16)(a2 * b2);
        }
#pragma unroll
        for (int k = 0; k < 9; ++k) atomicAdd(&rxz[e * 12 + k], pcd[k] * (1.f / 128.f));
    }
    __syncthreads();

    // ---- xz = LN(silu(zlin(rxz))) + LN(silu(zlin(rxz^T))) ----
    g1_pass(rxz, z_w1, z_b1, AB, 0, tid);
    __syncthreads();
    gemm_f16(AB, DD, ZW2h, DD, 48, 8, RAW, ZOUT, nullptr, 0, tid);
    __syncthreads();
    ln_apply(RAW, PROD, stats, z_b2, z_ln_g, z_ln_b, /*silu*/1, /*SET*/0, tid);

    g1_pass(rxz, z_w1, z_b1, AB, 1, tid);
    __syncthreads();
    gemm_f16(AB, DD, ZW2h, DD, 48, 8, RAW, ZOUT, nullptr, 0, tid);
    __syncthreads();
    ln_apply(RAW, PROD, stats, z_b2, z_ln_g, z_ln_b, 1, /*ADD*/1, tid);

    // ---- xij = LN(silu((xi*xj)@W1^T + b1) @ W2^T + b2) ----
    {
        const int si = sidx[e], di = sidx[EM + e];
        const float* xi = x + (size_t)si * DD + r * 64;
        const float* xj = x + (size_t)di * DD + r * 64;
        for (int o = 0; o < 64; ++o) AB[e * DD + r * 64 + o] = (f16)(xi[o] * xj[o]);
    }
    __syncthreads();
    gemm_f16(AB, DD, W1h, DD, 16, 8, RAW, DD, xij_b1, /*silu*/1, tid);  // h -> RAW (64x256)
    __syncthreads();
    {   // copy h into AB as next A-matrix
        const unsigned int* s = (const unsigned int*)RAW;
        unsigned int* d = (unsigned int*)AB;
        for (int i = tid; i < EM * DD / 2; i += TPB) d[i] = s[i];
    }
    __syncthreads();
    gemm_f16(AB, DD, W2h, DD, 48, 8, RAW, ZOUT, nullptr, 0, tid);
    __syncthreads();
    ln_apply(RAW, PROD, stats, xij_b2, xij_ln_g, xij_ln_b, 0, /*MUL*/2, tid);

    // ---- xzx = LN((Q @ ZX) @ zx_w^T + zx_b) ----
    gemm_f16(Q, 384, ZXth, 384, 16, 12, RAW, DD, nullptr, 0, tid);      // t -> RAW (64x256)
    __syncthreads();
    {
        const unsigned int* s = (const unsigned int*)RAW;
        unsigned int* d = (unsigned int*)AB;
        for (int i = tid; i < EM * DD / 2; i += TPB) d[i] = s[i];
    }
    __syncthreads();
    gemm_f16(AB, DD, ZXWh, DD, 48, 8, RAW, ZOUT, nullptr, 0, tid);
    __syncthreads();
    ln_apply(RAW, PROD, stats, zx_b, zx_ln_g, zx_ln_b, 0, /*MUL*/2, tid);

    // ---- out[e] = PROD row . lin_w + lin_b ----
    {
        float s = 0.f;
        const int c0 = r * 192;
        for (int i = 0; i < 192; ++i) s += (float)PROD[e * ZOUT + c0 + i] * lin_w[c0 + i];
        stats[e * 8 + r * 2] = s;
        __syncthreads();
        const int ge = e0 + e;
        if (r == 0 && ge < E_) {
            out[ge] = stats[e * 8] + stats[e * 8 + 2] + stats[e * 8 + 4] + stats[e * 8 + 6]
                      + lin_b[0];
        }
    }
}

// ---- ZX[c,l,d] = sum_n Zc[n,c,l] * x[n,d] (coeff applied at transpose) ----
__global__ __launch_bounds__(512) void zx_accum_kernel(const float* __restrict__ x,
                                                       const float* __restrict__ Z,
                                                       const float* __restrict__ zc_w,
                                                       const float* __restrict__ zc_b,
                                                       float* __restrict__ ZX, int N_, int chunk) {
    __shared__ float zrow[384];
    __shared__ float zc[128];
    const int tid = threadIdx.x;
    const int c   = blockIdx.y;
    const int n0  = blockIdx.x * chunk;
    const int n1  = min(n0 + chunk, N_);
    const int d = tid & 255, lh = tid >> 8;
    float acc[64];
#pragma unroll
    for (int j = 0; j < 64; ++j) acc[j] = 0.f;
    const float w0 = zc_w[c * 3 + 0], w1 = zc_w[c * 3 + 1], w2 = zc_w[c * 3 + 2], b = zc_b[c];
    for (int n = n0; n < n1; ++n) {
        __syncthreads();
        if (tid < 384) zrow[tid] = Z[(size_t)n * 384 + tid];
        const float xv = x[(size_t)n * 256 + d];
        __syncthreads();
        if (tid >= 384) {
            const int l = tid - 384;
            zc[l] = b + w0 * zrow[l] + w1 * zrow[128 + l] + w2 * zrow[256 + l];
        }
        __syncthreads();
        const float* zcp = zc + lh * 64;
#pragma unroll 16
        for (int j = 0; j < 64; ++j) acc[j] += zcp[j] * xv;
    }
    const int lbase = lh * 64;
    for (int j = 0; j < 64; ++j)
        atomicAdd(&ZX[(size_t)(c * 128 + lbase + j) * 256 + d], acc[j]);
}

__global__ void zero_kernel(float* p, int n) {
    int i = blockIdx.x * blockDim.x + threadIdx.x;
    if (i < n) p[i] = 0.f;
}

__global__ void cvt_kernel(const float* __restrict__ src, f16* __restrict__ dst, int n) {
    int i = blockIdx.x * blockDim.x + threadIdx.x;
    if (i < n) dst[i] = (f16)src[i];
}

// ZXt[d][c*128+l] = coeff * ZX[(c*128+l)*256 + d]
__global__ void zxt_kernel(const float* __restrict__ ZX, f16* __restrict__ ZXt, float coeff) {
    const int cl = blockIdx.x;
    const int d  = threadIdx.x;
    ZXt[(size_t)d * 384 + cl] = (f16)(ZX[(size_t)cl * 256 + d] * coeff);
}

extern "C" void kernel_launch(void* const* d_in, const int* in_sizes, int n_in,
                              void* d_out, int out_size, void* d_ws, size_t ws_size,
                              hipStream_t stream) {
    const float* x        = (const float*)d_in[0];
    const float* Z        = (const float*)d_in[1];
    const int*   tar_ei   = (const int*)d_in[3];
    const float* xij_w1   = (const float*)d_in[4];
    const float* xij_b1   = (const float*)d_in[5];
    const float* xij_w2   = (const float*)d_in[6];
    const float* xij_b2   = (const float*)d_in[7];
    const float* xij_ln_g = (const float*)d_in[8];
    const float* xij_ln_b = (const float*)d_in[9];
    const float* z_w1     = (const float*)d_in[10];
    const float* z_b1     = (const float*)d_in[11];
    const float* z_w2     = (const float*)d_in[12];
    const float* z_b2     = (const float*)d_in[13];
    const float* z_ln_g   = (const float*)d_in[14];
    const float* z_ln_b   = (const float*)d_in[15];
    const float* zc_w     = (const float*)d_in[16];
    const float* zc_b     = (const float*)d_in[17];
    const float* zx_w     = (const float*)d_in[18];
    const float* zx_b     = (const float*)d_in[19];
    const float* zx_ln_g  = (const float*)d_in[20];
    const float* zx_ln_b  = (const float*)d_in[21];
    const float* lin_w    = (const float*)d_in[22];
    const float* lin_b    = (const float*)d_in[23];

    const int N_ = in_sizes[0] / 256;
    const int E_ = in_sizes[3] / 2;

    char* ws   = (char*)d_ws;
    float* ZX  = (float*)(ws + 0);            // 384*256 f32
    f16* W1h   = (f16*)(ws + 393216);         // 256*256 f16
    f16* W2h   = (f16*)(ws + 524288);         // 768*256 f16
    f16* ZW2h  = (f16*)(ws + 917504);         // 768*256 f16
    f16* ZXWh  = (f16*)(ws + 1310720);        // 768*256 f16
    f16* ZXth  = (f16*)(ws + 1703936);        // 256*384 f16  (total 1900544 B)

    zero_kernel<<<384, 256, 0, stream>>>(ZX, 384 * 256);

    const int nsplit = 200;
    const int chunk  = (N_ + nsplit - 1) / nsplit;
    zx_accum_kernel<<<dim3(nsplit, 3), 512, 0, stream>>>(x, Z, zc_w, zc_b, ZX, N_, chunk);

    cvt_kernel<<<256, 256, 0, stream>>>(xij_w1, W1h, 256 * 256);
    cvt_kernel<<<768, 256, 0, stream>>>(xij_w2, W2h, 768 * 256);
    cvt_kernel<<<768, 256, 0, stream>>>(z_w2, ZW2h, 768 * 256);
    cvt_kernel<<<768, 256, 0, stream>>>(zx_w, ZXWh, 768 * 256);

    const float coeff = (float)(1.0 / (sqrt((double)N_) * 3840.0));
    zxt_kernel<<<384, 256, 0, stream>>>(ZX, ZXth, coeff);

    hipFuncSetAttribute((const void*)fused_edge_kernel,
                        hipFuncAttributeMaxDynamicSharedMemorySize, SMEM_BYTES);
    const int nblk = (E_ + EM - 1) / EM;
    fused_edge_kernel<<<nblk, TPB, SMEM_BYTES, stream>>>(
        x, Z, tar_ei, xij_b1, xij_b2, xij_ln_g, xij_ln_b, z_w1, z_b1, z_b2, z_ln_g, z_ln_b,
        zx_b, zx_ln_g, zx_ln_b, lin_w, lin_b, W1h, W2h, ZW2h, ZXWh, ZXth,
        (float*)d_out, E_, N_);
}
